// VectorQuantizer_21423296872766
// MI455X (gfx1250) — matrix operand order repaired and checked by reading, not compiled, not run on hardware
//
#include <hip/hip_runtime.h>
#include <stdint.h>

// ---------------------------------------------------------------------------
// VQ-VAE vector quantizer for MI455X (gfx1250, wave32, WMMA).
// Distance GEMM via V_WMMA_F32_16X16X32_BF16 with bf16 hi/lo split for
// fp32-class argmin accuracy. Deterministic segment sums (integer atomics).
// ---------------------------------------------------------------------------

typedef __bf16 v16bf __attribute__((ext_vector_type(16)));
typedef float  v8f   __attribute__((ext_vector_type(8)));

#define NTOK   16384u        // B*H*W = 16*32*32
#define KCODES 8192u
#define CDIM   64u
#define DECAYF 0.99f
#define OMDEC  0.01f         // (1-DECAY) rounded to f32 == float(0.01)
#define EPSV   1e-5f
#define FIXSC  16777216.0f   // 2^24 fixed-point scale for deterministic dw
#define INVFIX (1.0f/16777216.0f)

// ---- workspace layout (u32 units) -----------------------------------------
#define WS_A_HI   0u          // 1024 M-tiles * 2 kt * 32 lanes * 8 u32
#define WS_A_LO   524288u
#define WS_B_HI   1048576u    // 512 N-tiles * 2 kt * 32 lanes * 8 u32
#define WS_B_LO   1310720u
#define WS_ENORM  1572864u    // f32[8192]
#define WS_IDX    1581056u    // i32[16384]
#define WS_COUNTS 1597440u    // u32[8192]
#define WS_DW     1605632u    // i64[524288] (byte offset divisible by 8)
#define WS_NSUM   2654208u    // f32[1]
#define WS_ZERO_N (8192u + 1048576u + 1u)   // counts+dw+nsum contiguous

// ---- output layout (f32 units, reference tuple order) ----------------------
#define OUT_ZQST 0u
#define OUT_IDX  1048576u
#define OUT_ZQ   1064960u
#define OUT_EMB  2113536u
#define OUT_CS   2637824u
#define OUT_AVG  2646016u

// ---- helpers ---------------------------------------------------------------
__device__ __forceinline__ unsigned bf16_bits_rne(float x) {
    union { float f; unsigned u; } v; v.f = x;
    unsigned u = v.u;
    return ((u + 0x7FFFu + ((u >> 16) & 1u)) >> 16) & 0xFFFFu;
}
__device__ __forceinline__ float bf16_bits_to_float(unsigned b) {
    union { float f; unsigned u; } v; v.u = b << 16; return v.f;
}
__device__ __forceinline__ v8f wmma_bf16(v16bf a, v16bf b, v8f c) {
    // emits v_wmma_f32_16x16x32_bf16
    return __builtin_amdgcn_wmma_f32_16x16x32_bf16(
        false, a, false, b, (short)0, c, false, false);
}

// ---- kernel: zero the atomic accumulators ---------------------------------
__global__ void k_zero(unsigned* __restrict__ W) {
    unsigned i = blockIdx.x * 256u + threadIdx.x;
    if (i < WS_ZERO_N) W[WS_COUNTS + i] = 0u;
}

// ---- kernel: pack z_e (NCHW) into WMMA A-fragment hi/lo bf16 layout --------
// A 16x32 (MxK): lanes0-15 M=lane, V0-3: K=0..7(pairs), V4-7: K=16..23;
//                lanes16-31 M=lane-16, V0-3: K=8..15, V4-7: K=24..31.
__global__ void k_pack_a(const float* __restrict__ z_e, unsigned* __restrict__ W) {
    unsigned tid  = blockIdx.x * 256u + threadIdx.x;    // 524288 threads
    unsigned v    = tid & 7u;
    unsigned lane = (tid >> 3) & 31u;
    unsigned kt   = (tid >> 8) & 1u;
    unsigned t    = tid >> 9;                           // M-tile 0..1023
    unsigned m    = lane & 15u, half = lane >> 4;
    unsigned n    = t * 16u + m;
    unsigned b    = n >> 10, hw = n & 1023u;
    unsigned hi = 0u, lo = 0u;
#pragma unroll
    for (unsigned e = 0; e < 2u; ++e) {
        unsigned kl = (v >> 2) * 16u + half * 8u + (v & 3u) * 2u + e;
        unsigned c  = kt * 32u + kl;
        float x  = z_e[(b * 64u + c) * 1024u + hw];
        unsigned hb = bf16_bits_rne(x);
        unsigned lb = bf16_bits_rne(x - bf16_bits_to_float(hb));
        hi |= hb << (16u * e);
        lo |= lb << (16u * e);
    }
    unsigned off = ((t * 2u + kt) * 32u + lane) * 8u + v;
    W[WS_A_HI + off] = hi;
    W[WS_A_LO + off] = lo;
}

// ---- kernel: pack embedding into WMMA B-fragment hi/lo bf16 layout ---------
// B 32x16 (KxN): lane n=lane&15; lanes0-15 V0-3: K=0..7, V4-7: K=16..23; lanes16-31 V0-3: K=8..15, V4-7: K=24..31.
__global__ void k_pack_b(const float* __restrict__ emb, unsigned* __restrict__ W) {
    unsigned tid  = blockIdx.x * 256u + threadIdx.x;    // 262144 threads
    unsigned v    = tid & 7u;
    unsigned lane = (tid >> 3) & 31u;
    unsigned kt   = (tid >> 8) & 1u;
    unsigned j    = tid >> 9;                           // N-tile 0..511
    unsigned ncol = lane & 15u, half = lane >> 4;
    unsigned k    = j * 16u + ncol;
    unsigned hi = 0u, lo = 0u;
#pragma unroll
    for (unsigned e = 0; e < 2u; ++e) {
        unsigned kl = (v >> 2) * 16u + half * 8u + (v & 3u) * 2u + e;
        unsigned c  = kt * 32u + kl;
        float x  = emb[k * 64u + c];
        unsigned hb = bf16_bits_rne(x);
        unsigned lb = bf16_bits_rne(x - bf16_bits_to_float(hb));
        hi |= hb << (16u * e);
        lo |= lb << (16u * e);
    }
    unsigned off = ((j * 2u + kt) * 32u + lane) * 8u + v;
    W[WS_B_HI + off] = hi;
    W[WS_B_LO + off] = lo;
}

// ---- kernel: ||e_k||^2 in fp32 ---------------------------------------------
__global__ void k_enorm(const float* __restrict__ emb, unsigned* __restrict__ W) {
    unsigned k = blockIdx.x * 256u + threadIdx.x;
    if (k >= KCODES) return;
    float s = 0.f;
#pragma unroll 8
    for (unsigned c = 0; c < CDIM; ++c) { float e = emb[k * 64u + c]; s = fmaf(e, e, s); }
    ((float*)W)[WS_ENORM + k] = s;
}

// ---- kernel: distance GEMM (bf16x3 split WMMA) + argmin over K -------------
// One wave owns 32 tokens (two 16x16 C-tiles), streams all 512 code tiles.
__global__ void __launch_bounds__(256) k_dist_argmin(unsigned* __restrict__ W,
                                                     float* __restrict__ out) {
    const unsigned lane = threadIdx.x & 31u;
    const unsigned wid  = threadIdx.x >> 5;
    const unsigned gw   = blockIdx.x * 8u + wid;        // 0..511
    const v16bf* Ah = (const v16bf*)(W + WS_A_HI);
    const v16bf* Al = (const v16bf*)(W + WS_A_LO);
    const v16bf* Bh = (const v16bf*)(W + WS_B_HI);
    const v16bf* Bl = (const v16bf*)(W + WS_B_LO);
    const float* en = (const float*)(W + WS_ENORM);
    int* idxw = (int*)(W + WS_IDX);

    const unsigned tm0 = gw * 2u, tm1 = tm0 + 1u;
    // register-resident A fragments (hi/lo, 2 K-chunks, 2 M-tiles)
    v16bf ah00 = Ah[(tm0 * 2u + 0u) * 32u + lane];
    v16bf ah01 = Ah[(tm0 * 2u + 1u) * 32u + lane];
    v16bf al00 = Al[(tm0 * 2u + 0u) * 32u + lane];
    v16bf al01 = Al[(tm0 * 2u + 1u) * 32u + lane];
    v16bf ah10 = Ah[(tm1 * 2u + 0u) * 32u + lane];
    v16bf ah11 = Ah[(tm1 * 2u + 1u) * 32u + lane];
    v16bf al10 = Al[(tm1 * 2u + 0u) * 32u + lane];
    v16bf al11 = Al[(tm1 * 2u + 1u) * 32u + lane];

    float bv0[8], bv1[8];
    int   bi0[8], bi1[8];
#pragma unroll
    for (int v = 0; v < 8; ++v) { bv0[v] = 3.4e38f; bv1[v] = 3.4e38f; bi0[v] = 0; bi1[v] = 0; }

    const unsigned col = lane & 15u;
    for (unsigned j = 0; j < 512u; ++j) {
        if (j + 4u < 512u) {   // stream hint -> global_prefetch
            __builtin_prefetch((const void*)&Bh[((j + 4u) * 2u) * 32u + lane], 0, 0);
            __builtin_prefetch((const void*)&Bl[((j + 4u) * 2u) * 32u + lane], 0, 0);
        }
        v16bf bh0 = Bh[(j * 2u + 0u) * 32u + lane];
        v16bf bh1 = Bh[(j * 2u + 1u) * 32u + lane];
        v16bf bl0 = Bl[(j * 2u + 0u) * 32u + lane];
        v16bf bl1 = Bl[(j * 2u + 1u) * 32u + lane];
        float e = en[j * 16u + col];

        v8f a0 = {}; v8f a1 = {};
        a0 = wmma_bf16(ah00, bh0, a0);  a0 = wmma_bf16(ah01, bh1, a0);
        a0 = wmma_bf16(al00, bh0, a0);  a0 = wmma_bf16(al01, bh1, a0);
        a0 = wmma_bf16(ah00, bl0, a0);  a0 = wmma_bf16(ah01, bl1, a0);
        a1 = wmma_bf16(ah10, bh0, a1);  a1 = wmma_bf16(ah11, bh1, a1);
        a1 = wmma_bf16(al10, bh0, a1);  a1 = wmma_bf16(al11, bh1, a1);
        a1 = wmma_bf16(ah10, bl0, a1);  a1 = wmma_bf16(ah11, bl1, a1);

        const int kidx = (int)(j * 16u + col);
#pragma unroll
        for (int v = 0; v < 8; ++v) {
            float s0 = e - 2.0f * a0[v];
            if (s0 < bv0[v]) { bv0[v] = s0; bi0[v] = kidx; }
            float s1 = e - 2.0f * a1[v];
            if (s1 < bv1[v]) { bv1[v] = s1; bi1[v] = kidx; }
        }
    }

    // reduce (val,idx) across the 16 lanes of each half-group; first-index ties
#pragma unroll
    for (int off = 1; off < 16; off <<= 1) {
#pragma unroll
        for (int v = 0; v < 8; ++v) {
            float ov = __shfl_xor(bv0[v], off, 32);
            int   oi = __shfl_xor(bi0[v], off, 32);
            if (ov < bv0[v] || (ov == bv0[v] && oi < bi0[v])) { bv0[v] = ov; bi0[v] = oi; }
            ov = __shfl_xor(bv1[v], off, 32);
            oi = __shfl_xor(bi1[v], off, 32);
            if (ov < bv1[v] || (ov == bv1[v] && oi < bi1[v])) { bv1[v] = ov; bi1[v] = oi; }
        }
    }

    if (col == 0u) {           // lanes 0 and 16: rows m = half*8 + v
        unsigned half = lane >> 4;
#pragma unroll
        for (int v = 0; v < 8; ++v) {
            unsigned n0 = tm0 * 16u + half * 8u + (unsigned)v;
            unsigned n1 = tm1 * 16u + half * 8u + (unsigned)v;
            idxw[n0] = bi0[v];  out[OUT_IDX + n0] = (float)bi0[v];
            idxw[n1] = bi1[v];  out[OUT_IDX + n1] = (float)bi1[v];
        }
    }
}

// ---- kernel: gather z_q, z_q_st; deterministic counts/dw accumulation ------
__global__ void k_gather(const float* __restrict__ z_e, const float* __restrict__ emb,
                         unsigned* __restrict__ W, float* __restrict__ out) {
    unsigned n = blockIdx.x * 256u + threadIdx.x;
    if (n >= NTOK) return;
    int k = ((int*)(W + WS_IDX))[n];
    unsigned b = n >> 10, hw = n & 1023u;
    atomicAdd(&W[WS_COUNTS + (unsigned)k], 1u);                 // exact integer
    unsigned long long* dw = (unsigned long long*)(W + WS_DW);
#pragma unroll 4
    for (unsigned c = 0; c < CDIM; ++c) {
        float q = emb[(unsigned)k * 64u + c];
        unsigned g = (b * 64u + c) * 1024u + hw;
        float x = z_e[g];
        out[OUT_ZQST + g] = x + (q - x);   // straight-through, as computed by ref
        out[OUT_ZQ   + g] = q;
        long long fx = (long long)llrintf(x * FIXSC);           // 2^24 fixed point
        atomicAdd(&dw[(unsigned)k * 64u + c], (unsigned long long)fx);
    }
}

// ---- kernel: EMA updates ---------------------------------------------------
__global__ void k_ema_a(const float* __restrict__ cluster, const float* __restrict__ avg,
                        unsigned* __restrict__ W, float* __restrict__ out) {
    unsigned i = blockIdx.x * 256u + threadIdx.x;
    if (i >= KCODES * CDIM) return;
    long long d = (long long)((unsigned long long*)(W + WS_DW))[i];
    float dwf = (float)d * INVFIX;
    out[OUT_AVG + i] = DECAYF * avg[i] + OMDEC * dwf;
    if (i < KCODES) {
        float ncs = DECAYF * cluster[i] + OMDEC * (float)W[WS_COUNTS + i];
        out[OUT_CS + i] = ncs;
    }
}

// fixed-order single-block reduction: deterministic n = sum(new_cluster_size)
__global__ void k_reduce(const float* __restrict__ out, unsigned* __restrict__ W) {
    __shared__ float s[256];
    float acc = 0.f;
    for (unsigned t = threadIdx.x; t < KCODES; t += 256u) acc += out[OUT_CS + t];
    s[threadIdx.x] = acc;
    __syncthreads();
    for (int o = 128; o > 0; o >>= 1) {
        if ((int)threadIdx.x < o) s[threadIdx.x] += s[threadIdx.x + o];
        __syncthreads();
    }
    if (threadIdx.x == 0) ((float*)W)[WS_NSUM] = s[0];
}

__global__ void k_ema_b(unsigned* __restrict__ W, float* __restrict__ out) {
    unsigned i = blockIdx.x * 256u + threadIdx.x;
    if (i >= KCODES * CDIM) return;
    float nsum = ((float*)W)[WS_NSUM];
    unsigned k = i >> 6;
    float ncs = out[OUT_CS + k];
    float cs  = (ncs + EPSV) / (nsum + (float)KCODES * EPSV) * nsum;
    out[OUT_EMB + i] = out[OUT_AVG + i] / cs;
}

// ---------------------------------------------------------------------------
extern "C" void kernel_launch(void* const* d_in, const int* in_sizes, int n_in,
                              void* d_out, int out_size, void* d_ws, size_t ws_size,
                              hipStream_t stream) {
    const float* z_e     = (const float*)d_in[0];   // (16,64,32,32)
    const float* emb     = (const float*)d_in[1];   // (8192,64)
    const float* cluster = (const float*)d_in[2];   // (8192,)
    const float* avg     = (const float*)d_in[3];   // (8192,64)
    float*    out = (float*)d_out;
    unsigned* W   = (unsigned*)d_ws;
    (void)in_sizes; (void)n_in; (void)out_size; (void)ws_size;

    k_zero       <<<(WS_ZERO_N + 255u) / 256u, 256, 0, stream>>>(W);
    k_pack_a     <<<2048, 256, 0, stream>>>(z_e, W);
    k_pack_b     <<<1024, 256, 0, stream>>>(emb, W);
    k_enorm      <<<  32, 256, 0, stream>>>(emb, W);
    k_dist_argmin<<<  64, 256, 0, stream>>>(W, out);
    k_gather     <<<  64, 256, 0, stream>>>(z_e, emb, W, out);
    k_ema_a      <<<2048, 256, 0, stream>>>(cluster, avg, W, out);
    k_reduce     <<<   1, 256, 0, stream>>>(out, W);
    k_ema_b      <<<2048, 256, 0, stream>>>(W, out);
}
